// ScaledDotProductAttention_51616916963548
// MI455X (gfx1250) — compile-verified, matrix operand order checked by reading, not run on hardware
//
#include <hip/hip_runtime.h>
#include <math.h>

typedef __attribute__((ext_vector_type(16))) _Float16 v16h;
typedef __attribute__((ext_vector_type(8)))  float    v8f;
typedef __attribute__((ext_vector_type(4)))  float    f32x4;

#define B_  16
#define S_  4096
#define D_  64
#define WAVES_PER_BLOCK 4   // 4 waves * 32 q-rows = 128 q-rows per block

// ---------------------------------------------------------------------------
// Fragment element -> K mapping for 16-bit A (16x32) and B (32x16) operands
// (wave32): element j of the per-lane v16h maps to
//   k = (j&8 ? 16 : 0) + (lane>>4)*8 + (j&7)
// A: lanes hold row M = lane&15.  B: lanes hold column N = lane&15.
// C/D f32 16x16: lane holds col N=lane&15; VGPR r holds row M = r + 8*(lane>>4)
// ---------------------------------------------------------------------------

#define WMMA_F16(A, Bf, C) \
    __builtin_amdgcn_wmma_f32_16x16x32_f16(false, (A), false, (Bf), (short)0, (C), false, false)

// Load a 16-element f16 fragment from one row of a row-major f32 matrix.
// Columns form two aligned 8-float chunks: [cbase+h*8 ..] and [cbase+16+h*8 ..]
__device__ inline v16h load_frag_row(const float* __restrict__ rowptr, int cbase, int h) {
    const f32x4* p0 = (const f32x4*)(rowptr + cbase + h * 8);
    const f32x4* p1 = (const f32x4*)(rowptr + cbase + 16 + h * 8);
    f32x4 a0 = p0[0], a1 = p0[1];
    f32x4 b0 = p1[0], b1 = p1[1];
    v16h r;
#pragma unroll
    for (int j = 0; j < 4; ++j) {
        r[j]      = (_Float16)a0[j];
        r[4 + j]  = (_Float16)a1[j];
        r[8 + j]  = (_Float16)b0[j];
        r[12 + j] = (_Float16)b1[j];
    }
    return r;
}

// B-fragment of V: vcol points at V[(b*S + kc)*D + nd*16 + n] (n = lane&15).
__device__ inline v16h load_vfrag(const float* __restrict__ vcol, int h) {
    v16h r;
#pragma unroll
    for (int j = 0; j < 16; ++j) {
        int k = ((j & 8) ? 16 : 0) + h * 8 + (j & 7);
        r[j] = (_Float16)vcol[(size_t)k * D_];
    }
    return r;
}

// Per-lane online softmax update (no cross-lane traffic in the hot loop).
__device__ inline void online_update(float s, float& m, float& l) {
    float tm = fmaxf(m, s);
    l = fmaf(l, __expf(m - tm), __expf(s - tm));
    m = tm;
}

// Cross-lane reduce over the 16-lane half (xor bits 0..3) — used ONCE per wave.
__device__ inline float redmax16(float x) {
    x = fmaxf(x, __shfl_xor(x, 1));
    x = fmaxf(x, __shfl_xor(x, 2));
    x = fmaxf(x, __shfl_xor(x, 4));
    x = fmaxf(x, __shfl_xor(x, 8));
    return x;
}
__device__ inline float redsum16(float x) {
    x += __shfl_xor(x, 1);
    x += __shfl_xor(x, 2);
    x += __shfl_xor(x, 4);
    x += __shfl_xor(x, 8);
    return x;
}

__global__ __launch_bounds__(WAVES_PER_BLOCK * 32)
void sdpa_wmma_kernel(const float* __restrict__ q,
                      const float* __restrict__ k,
                      const float* __restrict__ v,
                      float* __restrict__ out_ctx,
                      float* __restrict__ out_att) {
    // Per-wave P staging buffer (f16): 32 q-rows x 32 k-cols, padded rows.
    __shared__ _Float16 Pbuf[WAVES_PER_BLOCK][32][40];

    const int wave = threadIdx.x >> 5;
    const int lane = threadIdx.x & 31;
    const int m    = lane & 15;   // A row / B col / C col index
    const int h    = lane >> 4;   // half-wave selector

    const int qb = blockIdx.x & 31;        // S_/128 = 32 q-blocks
    const int b  = blockIdx.x >> 5;
    const int q0 = qb * 128 + wave * 32;   // first q row of this wave's 2 tiles

    const float scale = 0.125f;            // 1/sqrt(64)

    // ---- Load Q A-fragments once: tile0 rows q0+m, tile1 rows q0+16+m ----
    const float* qrow0 = q + ((size_t)(b * S_ + q0 + m)) * D_;
    const float* qrow1 = qrow0 + (size_t)16 * D_;
    const v16h qa0 = load_frag_row(qrow0, 0, h);
    const v16h qa1 = load_frag_row(qrow0, 32, h);
    const v16h qa2 = load_frag_row(qrow1, 0, h);
    const v16h qa3 = load_frag_row(qrow1, 32, h);

    const float* kbase = k + (size_t)b * S_ * D_;

    // ---- Pass A: per-lane online softmax stats; each K fragment pair feeds
    //      4 WMMAs (2 q-tiles x 2 d-steps) -> 2x K-load reuse. ----
    float m0[8], l0[8], m1[8], l1[8];
#pragma unroll
    for (int r = 0; r < 8; ++r) {
        m0[r] = -__builtin_inff(); l0[r] = 0.0f;
        m1[r] = -__builtin_inff(); l1[r] = 0.0f;
    }

    for (int kb = 0; kb < S_ / 16; ++kb) {
        const float* krow = kbase + (size_t)(kb * 16 + m) * D_;
        v16h kf0 = load_frag_row(krow, 0, h);
        v16h kf1 = load_frag_row(krow, 32, h);
        v8f c0 = {}, c1 = {};
        c0 = WMMA_F16(qa0, kf0, c0);
        c0 = WMMA_F16(qa1, kf1, c0);
        c1 = WMMA_F16(qa2, kf0, c1);
        c1 = WMMA_F16(qa3, kf1, c1);
#pragma unroll
        for (int r = 0; r < 8; ++r) {
            online_update(c0[r] * scale, m0[r], l0[r]);
            online_update(c1[r] * scale, m1[r], l1[r]);
        }
    }

    // ---- Single split-softmax merge + fold into z = m + log(l):
    //      p = exp(s - z) is fully normalized. ----
    float z0[8], z1[8];
#pragma unroll
    for (int r = 0; r < 8; ++r) {
        float nm = redmax16(m0[r]);
        float ls = redsum16(l0[r] * __expf(m0[r] - nm));
        z0[r] = nm + __logf(ls);
        nm = redmax16(m1[r]);
        ls = redsum16(l1[r] * __expf(m1[r] - nm));
        z1[r] = nm + __logf(ls);
    }

    // ---- Pass B: recompute scores, emit normalized attention, P @ V.
    //      Each V fragment gather now feeds 2 PV WMMAs. ----
    v8f acc0[4] = {}, acc1[4] = {};          // 32 x 64 f32 context accumulators
    float* att_base = out_att + ((size_t)(b * S_ + q0)) * S_;
    const float* vbase = v + (size_t)b * S_ * D_;

    for (int kc = 0; kc < S_; kc += 32) {
#pragma unroll
        for (int t = 0; t < 2; ++t) {
            const int kb16 = kc + t * 16;
            const float* krow = kbase + (size_t)(kb16 + m) * D_;
            v16h kf0 = load_frag_row(krow, 0, h);
            v16h kf1 = load_frag_row(krow, 32, h);
            v8f c0 = {}, c1 = {};
            c0 = WMMA_F16(qa0, kf0, c0);
            c0 = WMMA_F16(qa1, kf1, c0);
            c1 = WMMA_F16(qa2, kf0, c1);
            c1 = WMMA_F16(qa3, kf1, c1);
#pragma unroll
            for (int r = 0; r < 8; ++r) {
                const int row = r + 8 * h;
                float p = __expf(fmaf(c0[r], scale, -z0[r]));      // normalized
                att_base[(size_t)row * S_ + kb16 + m] = p;
                Pbuf[wave][row][t * 16 + m] = (_Float16)p;
                float p1 = __expf(fmaf(c1[r], scale, -z1[r]));
                att_base[(size_t)(row + 16) * S_ + kb16 + m] = p1;
                Pbuf[wave][row + 16][t * 16 + m] = (_Float16)p1;
            }
        }
        // Wave-local LDS: DS ops from one wave complete in order; the wait +
        // memory clobber keeps the compiler from hoisting the reads.
        asm volatile("s_wait_dscnt 0" ::: "memory");

        // Reload P as A-fragments (16 x 32 each, K = kc..kc+31)
        v16h pa0, pa1;
#pragma unroll
        for (int j = 0; j < 16; ++j) {
            int kk = ((j & 8) ? 16 : 0) + h * 8 + (j & 7);
            pa0[j] = Pbuf[wave][m][kk];
            pa1[j] = Pbuf[wave][m + 16][kk];
        }

        const float* vtile = vbase + (size_t)kc * D_;
#pragma unroll
        for (int nd = 0; nd < 4; ++nd) {
            v16h vf = load_vfrag(vtile + nd * 16 + m, h);
            acc0[nd] = WMMA_F16(pa0, vf, acc0[nd]);
            acc1[nd] = WMMA_F16(pa1, vf, acc1[nd]);
        }
    }

    // ---- Write context (already normalized) ----
    float* ctx = out_ctx + ((size_t)(b * S_ + q0)) * D_;
#pragma unroll
    for (int nd = 0; nd < 4; ++nd) {
#pragma unroll
        for (int r = 0; r < 8; ++r) {
            const int row = r + 8 * h;
            ctx[(size_t)row * D_ + nd * 16 + m] = acc0[nd][r];
            ctx[(size_t)(row + 16) * D_ + nd * 16 + m] = acc1[nd][r];
        }
    }
}

extern "C" void kernel_launch(void* const* d_in, const int* in_sizes, int n_in,
                              void* d_out, int out_size, void* d_ws, size_t ws_size,
                              hipStream_t stream) {
    const float* q = (const float*)d_in[0];
    const float* k = (const float*)d_in[1];
    const float* v = (const float*)d_in[2];
    float* out     = (float*)d_out;
    float* out_ctx = out;                                 // [B,S,D] first
    float* out_att = out + (size_t)B_ * S_ * D_;          // then [B,S,S]

    dim3 grid(B_ * (S_ / 128));      // 512 workgroups
    dim3 block(WAVES_PER_BLOCK * 32);
    hipLaunchKernelGGL(sdpa_wmma_kernel, grid, block, 0, stream,
                       q, k, v, out_ctx, out_att);
}
